// Graph_14551349199223
// MI455X (gfx1250) — matrix operand order repaired and checked by reading, not compiled, not run on hardware
//
#include <hip/hip_runtime.h>
#include <hip/hip_bf16.h>
#include <math.h>

// ---------------------------------------------------------------------------
// Model dims
// ---------------------------------------------------------------------------
#define Bsz   128
#define BUFF  64
#define STK   48
#define NCOMP 8
#define HIST  96
#define WDIM  300
#define POSD  50
#define FC    128
#define UU    256
#define NOUT  82
#define NOUTP 96      // NOUT padded to 16
#define EKP   352     // 350 padded to 32
#define RECK  306

typedef _Float16 h16;
typedef __attribute__((ext_vector_type(16))) _Float16 v16h;
typedef __attribute__((ext_vector_type(8)))  _Float16 v8h;
typedef __attribute__((ext_vector_type(8)))  float    v8f;

#if __has_builtin(__builtin_amdgcn_sched_barrier)
#define SCHED_FENCE() __builtin_amdgcn_sched_barrier(0)
#else
#define SCHED_FENCE() asm volatile("" ::: "memory")
#endif

__device__ __forceinline__ float sigf(float x) { return 1.0f / (1.0f + __expf(-x)); }

// WMMA: D = A(16x32 f16) x B(32x16 f16) + C(16x16 f32)
__device__ __forceinline__ v8f wmma32(v16h a, v16h b, v8f c) {
  return __builtin_amdgcn_wmma_f32_16x16x32_f16(false, a, false, b, (short)0, c,
                                                false, false);
}

// Load one lane's 16 contiguous halves (32B) as two 128-bit loads.
__device__ __forceinline__ v16h ld_frag(const h16* p) {
  v8h lo = *(const v8h*)p;
  v8h hi = *(const v8h*)(p + 8);
  return __builtin_shufflevector(lo, hi, 0, 1, 2, 3, 4, 5, 6, 7,
                                 8, 9, 10, 11, 12, 13, 14, 15);
}

// ---------------------------------------------------------------------------
// Fragment-packed layouts.
// A layout (CDNA5 16-bit A, 16x32 per chunk): lane = hi*16 + m,
//   element e = 2i+s with k_in_chunk = (i<4 ? 2i : 16+2(i-4)) + 8*hi + s.
// Inverse (m,k) -> flat packed index:
__device__ __forceinline__ int a_pack_idx(int m, int k) {
  int kc = k >> 5, kr = k & 31;
  int hi = (kr >> 3) & 1;
  int i  = ((kr & 7) >> 1) + 4 * (kr >> 4);
  int s  = kr & 1;
  return ((kc * 32) + hi * 16 + m) * 16 + (2 * i + s);
}
// Packed A fragment load from LDS: chunk kc, this lane.
__device__ __forceinline__ v16h load_a_pk(const h16* A, int kc) {
  return ld_frag(A + ((size_t)(kc * 32 + (threadIdx.x & 31))) * 16);
}
// Packed B fragment load from global: chunk kc, n-tile nt, nnt tiles total.
__device__ __forceinline__ v16h load_b_pk(const h16* Wp, int nnt, int kc, int nt) {
  return ld_frag(Wp + (((size_t)kc * nnt + nt) * 32 + (threadIdx.x & 31)) * 16);
}

// ---------------------------------------------------------------------------
// One 16xN-tile GEMM chain over NK K-chunks.  B fragments are loaded in
// groups of GS, pinned above their WMMA consumers by a scheduling barrier so
// the whole group stays in flight together (memory-level parallelism).
// ---------------------------------------------------------------------------
template <int NK, int GS>
__device__ __forceinline__ v8f gemm_tile(const h16* Apk, const h16* Wp,
                                         int nnt, int nt) {
  v8f acc = {};
#pragma unroll
  for (int g = 0; g < (NK + GS - 1) / GS; ++g) {
    v16h bf[GS];
#pragma unroll
    for (int j = 0; j < GS; ++j)
      if (g * GS + j < NK) bf[j] = load_b_pk(Wp, nnt, g * GS + j, nt);
    SCHED_FENCE();
#pragma unroll
    for (int j = 0; j < GS; ++j)
      if (g * GS + j < NK) acc = wmma32(load_a_pk(Apk, g * GS + j), bf[j], acc);
  }
  return acc;
}

// ---------------------------------------------------------------------------
// Pack f32 weights (K,N) into fragment-major f16, padded to (Kp,Np).
// B layout: lane = hi*16 + n, element e  <->  k = kc*32 + 8*hi + (e&7) + 16*(e>>3).
// ---------------------------------------------------------------------------
__global__ void k_pack(const float* __restrict__ src, h16* __restrict__ dst,
                       int K, int N, int Kp, int Np) {
  int i = blockIdx.x * blockDim.x + threadIdx.x;
  if (i >= Kp * Np) return;
  int nnt  = Np >> 4;
  int e    = i & 15;
  int lane = (i >> 4) & 31;
  int tile = i >> 9;
  int nt = tile % nnt, kc = tile / nnt;
  int hi = lane >> 4, n = lane & 15;
  int ii = e >> 1, s = e & 1;
  int k   = kc * 32 + 8 * hi + (e & 7) + 16 * (e >> 3);
  int col = nt * 16 + n;
  float v = (k < K && col < N) ? src[k * N + col] : 0.0f;
  dst[i] = (h16)v;
}

// ---------------------------------------------------------------------------
// Embedding: relu(concat(w_emb[wid], p_emb[pid]) @ emb_W + emb_b)
// One wave per 16-token M tile; A staged packed in LDS; grouped-MLP WMMAs.
// ---------------------------------------------------------------------------
__global__ void k_embed(const int* __restrict__ bw, const int* __restrict__ bp,
                        const int* __restrict__ cw, const int* __restrict__ cp,
                        const float* __restrict__ w_emb, const float* __restrict__ p_emb,
                        const h16* __restrict__ embWp, const float* __restrict__ emb_b,
                        float* __restrict__ buffEmb, float* __restrict__ compEmb) {
  __shared__ h16 As[4][16 * EKP];     // packed A per wave
  __shared__ int ids[4][32];
  int wave = threadIdx.x >> 5, lane = threadIdx.x & 31;
  int mt = blockIdx.x * 4 + wave;     // 0..6655
  int g0 = mt * 16;

  if (lane < 16) {
    int g = g0 + lane, w, p;
    if (g < Bsz * BUFF) { w = bw[g]; p = bp[g]; }
    else { int gc = g - Bsz * BUFF; w = cw[gc]; p = cp[gc]; }
    ids[wave][lane] = w;
    ids[wave][16 + lane] = p;
  }
  __syncthreads();

  for (int idx = lane; idx < 16 * EKP; idx += 32) {
    int r = idx / EKP, k = idx % EKP;
    float v;
    if (k < WDIM)             v = w_emb[(size_t)ids[wave][r] * WDIM + k];
    else if (k < WDIM + POSD) v = p_emb[ids[wave][16 + r] * POSD + (k - WDIM)];
    else                      v = 0.0f;
    As[wave][a_pack_idx(r, k)] = (h16)v;
  }
  __syncthreads();

  float* outp = (g0 < Bsz * BUFF) ? (buffEmb + (size_t)g0 * FC)
                                  : (compEmb + (size_t)(g0 - Bsz * BUFF) * FC);
  int hi = lane >> 4, nl = lane & 15;
  for (int nt = 0; nt < 8; ++nt) {
    v8f acc = gemm_tile<11, 6>(As[wave], embWp, 8, nt);
    int col = nt * 16 + nl;
    float bb = emb_b[col];
#pragma unroll
    for (int r = 0; r < 8; ++r) {
      int row = hi ? r + 8 : r;
      float v = acc[r] + bb;
      outp[(size_t)row * FC + col] = v > 0.0f ? v : 0.0f;
    }
  }
}

// ---------------------------------------------------------------------------
// Compose: data-dependent 8-step recursion per (b,stk); one wave per sequence.
// ---------------------------------------------------------------------------
__global__ void k_compose(const int* __restrict__ cw, const int* __restrict__ ca,
                          const int* __restrict__ cal,
                          const float* __restrict__ compEmb,
                          const float* __restrict__ comp_a_emb,
                          const float* __restrict__ rec_W, const float* __restrict__ rec_b,
                          float* __restrict__ stackEmb) {
  __shared__ float vals[4][NCOMP][FC];
  int wave = threadIdx.x >> 5, lane = threadIdx.x & 31;
  int seq = blockIdx.x * 4 + wave;                 // 0..6143
  const int* wi = cw + seq * 2 * NCOMP;
  const float* ce = compEmb + (size_t)seq * 2 * NCOMP * FC;

  int head[NCOMP], dep[NCOMP], act[NCOMP];
#pragma unroll
  for (int n = 0; n < NCOMP; ++n) {
    head[n] = wi[2 * n];
    dep[n]  = wi[2 * n + 1];
    act[n]  = ca[seq * NCOMP + n];
  }

  for (int n = 0; n < NCOMP; ++n) {
    int ih = -1, idd = -1;
    for (int j = 0; j < n; ++j) {
      if (head[j] == head[n]) ih = j;
      if (head[j] == dep[n])  idd = j;
    }
    float acc0 = 0.f, acc1 = 0.f, acc2 = 0.f, acc3 = 0.f;
    for (int k = 0; k < RECK; ++k) {
      float xk;
      if (k < FC)           xk = (ih >= 0) ? vals[wave][ih][k] : ce[(2 * n) * FC + k];
      else if (k < FC + 50) xk = comp_a_emb[act[n] * 50 + (k - FC)];
      else                  xk = (idd >= 0) ? vals[wave][idd][k - FC - 50]
                                            : ce[(2 * n + 1) * FC + (k - FC - 50)];
      const float* wr = rec_W + k * FC + lane;
      acc0 = fmaf(xk, wr[0],  acc0);
      acc1 = fmaf(xk, wr[32], acc1);
      acc2 = fmaf(xk, wr[64], acc2);
      acc3 = fmaf(xk, wr[96], acc3);
    }
    vals[wave][n][lane]      = tanhf(acc0 + rec_b[lane]);
    vals[wave][n][lane + 32] = tanhf(acc1 + rec_b[lane + 32]);
    vals[wave][n][lane + 64] = tanhf(acc2 + rec_b[lane + 64]);
    vals[wave][n][lane + 96] = tanhf(acc3 + rec_b[lane + 96]);
    __syncthreads();
  }

  int al = cal[seq];
  int pick = al - 1; if (pick > NCOMP - 1) pick = NCOMP - 1; if (pick < 0) pick = 0;
  for (int q = 0; q < 4; ++q) {
    int j = lane + 32 * q;
    stackEmb[(size_t)seq * FC + j] = (al == 0) ? ce[j] : vals[wave][pick][j];
  }
}

// ---------------------------------------------------------------------------
// 2-layer LSTM, time loop in-kernel.  8 blocks x 256 threads (8 waves).
// Templated on (MODE, KX); per-step WMMA GEMMs use the grouped-MLP tile chain.
// Packed A in LDS ([x|h0], [h0|h1]); c state in registers.
// ---------------------------------------------------------------------------
template <int MODE, int KX>
__global__ void k_lstm(const float* __restrict__ xsrc, const int* __restrict__ hist_ids,
                       const float* __restrict__ hist_tbl, int T,
                       const h16* __restrict__ W0p, const float* __restrict__ b0,
                       const h16* __restrict__ W1p, const float* __restrict__ b1,
                       const int* __restrict__ lens, float* __restrict__ hout) {
  extern __shared__ char smem[];
  constexpr int K0  = KX + UU;
  constexpr int NK0 = K0 >> 5;
  h16*   A0 = (h16*)smem;               // packed, 16*K0 halves  ([x | h0])
  h16*   A1 = A0 + 16 * K0;             // packed, 16*512 halves ([h0 | h1])
  float* Z  = (float*)(A1 + 16 * 512);  // 16 x 1024 gate pre-activations

  int tid = threadIdx.x, wave = tid >> 5, lane = tid & 31;
  int tile = blockIdx.x;                // batch tile 0..7

  float c0r[16], c1r[16];
#pragma unroll
  for (int q = 0; q < 16; ++q) { c0r[q] = 0.f; c1r[q] = 0.f; }
  for (int i = tid; i < 16 * K0; i += 256)  A0[i] = (h16)0.f;
  for (int i = tid; i < 16 * 512; i += 256) A1[i] = (h16)0.f;
  __syncthreads();

  for (int t = 0; t < T; ++t) {
    // stage x_t (f32 -> f16, packed) for 16 samples
    for (int i = tid; i < 16 * KX; i += 256) {
      int r = i / KX, k = i % KX;
      int gs = tile * 16 + r;
      float v;
      if (MODE == 2) v = (k < 50) ? hist_tbl[hist_ids[gs * T + t] * 50 + k] : 0.0f;
      else           v = xsrc[((size_t)gs * T + t) * FC + k];
      A0[a_pack_idx(r, k)] = (h16)v;
    }
    __syncthreads();

    // layer 0 GEMM: 16 x 1024, K = K0
    for (int nq = 0; nq < 8; ++nq) {
      int nt = wave * 8 + nq;
      v8f acc = gemm_tile<NK0, 6>(A0, W0p, 64, nt);
      int col = nt * 16 + (lane & 15), hi = lane >> 4;
      float bb = b0[col];
#pragma unroll
      for (int r = 0; r < 8; ++r) {
        int row = hi ? r + 8 : r;
        Z[row * 1024 + col] = acc[r] + bb;
      }
    }
    __syncthreads();

    // layer 0 cell update (thread tid owns unit u=tid across 16 samples)
#pragma unroll
    for (int q = 0; q < 16; ++q) {
      int gs = tile * 16 + q, u = tid;
      if (t < lens[gs]) {
        float zi = Z[q * 1024 + u],       zj = Z[q * 1024 + 256 + u];
        float zf = Z[q * 1024 + 512 + u], zo = Z[q * 1024 + 768 + u];
        float c  = c0r[q] * sigf(zf + 1.0f) + sigf(zi) * tanhf(zj);
        float h  = tanhf(c) * sigf(zo);
        c0r[q] = c;
        A0[a_pack_idx(q, KX + u)] = (h16)h;
        A1[a_pack_idx(q, u)]      = (h16)h;
      }
    }
    __syncthreads();

    // layer 1 GEMM: 16 x 1024, K = 512
    for (int nq = 0; nq < 8; ++nq) {
      int nt = wave * 8 + nq;
      v8f acc = gemm_tile<16, 8>(A1, W1p, 64, nt);
      int col = nt * 16 + (lane & 15), hi = lane >> 4;
      float bb = b1[col];
#pragma unroll
      for (int r = 0; r < 8; ++r) {
        int row = hi ? r + 8 : r;
        Z[row * 1024 + col] = acc[r] + bb;
      }
    }
    __syncthreads();

#pragma unroll
    for (int q = 0; q < 16; ++q) {
      int gs = tile * 16 + q, u = tid;
      if (t < lens[gs]) {
        float zi = Z[q * 1024 + u],       zj = Z[q * 1024 + 256 + u];
        float zf = Z[q * 1024 + 512 + u], zo = Z[q * 1024 + 768 + u];
        float c  = c1r[q] * sigf(zf + 1.0f) + sigf(zi) * tanhf(zj);
        float h  = tanhf(c) * sigf(zo);
        c1r[q] = c;
        A1[a_pack_idx(q, 256 + u)] = (h16)h;
      }
    }
    __syncthreads();
  }

#pragma unroll
  for (int q = 0; q < 16; ++q) {
    int gs = tile * 16 + q;
    hout[(size_t)gs * UU + tid] = (float)A1[a_pack_idx(q, 256 + tid)];
  }
}

// ---------------------------------------------------------------------------
// Final projection: concat(hS,hB,hA) (128x768) @ fW (768x82) + fb
// One wave per 16-sample M tile; A staged packed in LDS.
// ---------------------------------------------------------------------------
__global__ void k_final(const float* __restrict__ hS, const float* __restrict__ hB,
                        const float* __restrict__ hA, const h16* __restrict__ fWp,
                        const float* __restrict__ fb, float* __restrict__ out) {
  __shared__ h16 As[16 * 768];          // packed A for this wave's M tile
  int lane = threadIdx.x & 31;
  int hi = lane >> 4;

  for (int idx = lane; idx < 16 * 768; idx += 32) {
    int r = idx / 768, k = idx % 768;
    int gs = blockIdx.x * 16 + r;
    float v = (k < 256) ? hS[gs * UU + k]
            : (k < 512) ? hB[gs * UU + k - 256]
                        : hA[gs * UU + k - 512];
    As[a_pack_idx(r, k)] = (h16)v;
  }
  __syncthreads();

  for (int nt = 0; nt < NOUTP / 16; ++nt) {
    v8f acc = gemm_tile<24, 6>(As, fWp, NOUTP / 16, nt);
    int col = nt * 16 + (lane & 15);
#pragma unroll
    for (int r = 0; r < 8; ++r) {
      int row = hi ? r + 8 : r;
      if (col < NOUT)
        out[(blockIdx.x * 16 + row) * NOUT + col] = acc[r] + fb[col];
    }
  }
}

// ---------------------------------------------------------------------------
extern "C" void kernel_launch(void* const* d_in, const int* in_sizes, int n_in,
                              void* d_out, int out_size, void* d_ws, size_t ws_size,
                              hipStream_t stream) {
  (void)in_sizes; (void)n_in; (void)out_size; (void)ws_size;
  const int* bw   = (const int*)d_in[0];
  const int* bp   = (const int*)d_in[1];
  const int* cw   = (const int*)d_in[2];
  const int* cp   = (const int*)d_in[3];
  const int* ca   = (const int*)d_in[4];
  const int* cal  = (const int*)d_in[5];
  const int* ha   = (const int*)d_in[6];
  const int* slen = (const int*)d_in[7];
  const int* blen = (const int*)d_in[8];
  const int* alen = (const int*)d_in[9];
  const float* p_emb      = (const float*)d_in[10];
  const float* comp_a_emb = (const float*)d_in[11];
  const float* hist_a_emb = (const float*)d_in[12];
  const float* w_emb      = (const float*)d_in[13];
  const float* emb_W = (const float*)d_in[14]; const float* emb_b = (const float*)d_in[15];
  const float* rec_W = (const float*)d_in[16]; const float* rec_b = (const float*)d_in[17];
  const float* sW0 = (const float*)d_in[18]; const float* sb0 = (const float*)d_in[19];
  const float* sW1 = (const float*)d_in[20]; const float* sb1 = (const float*)d_in[21];
  const float* bW0 = (const float*)d_in[22]; const float* bb0 = (const float*)d_in[23];
  const float* bW1 = (const float*)d_in[24]; const float* bb1 = (const float*)d_in[25];
  const float* aW0 = (const float*)d_in[26]; const float* ab0 = (const float*)d_in[27];
  const float* aW1 = (const float*)d_in[28]; const float* ab1 = (const float*)d_in[29];
  const float* fW  = (const float*)d_in[30]; const float* fb  = (const float*)d_in[31];
  float* out = (float*)d_out;

  char* ws = (char*)d_ws; size_t off = 0;
  auto carve = [&](size_t b) -> char* {
    char* p = ws + off; off = (off + b + 255) & ~(size_t)255; return p;
  };
  float* buffEmb  = (float*)carve((size_t)Bsz * BUFF * FC * 4);
  float* compEmb  = (float*)carve((size_t)Bsz * STK * 16 * FC * 4);
  float* stackEmb = (float*)carve((size_t)Bsz * STK * FC * 4);
  float* hS = (float*)carve((size_t)Bsz * UU * 4);
  float* hB = (float*)carve((size_t)Bsz * UU * 4);
  float* hA = (float*)carve((size_t)Bsz * UU * 4);
  h16* embWp = (h16*)carve((size_t)EKP * FC * 2);
  h16* sW0p  = (h16*)carve((size_t)384 * 1024 * 2);
  h16* sW1p  = (h16*)carve((size_t)512 * 1024 * 2);
  h16* bW0p  = (h16*)carve((size_t)384 * 1024 * 2);
  h16* bW1p  = (h16*)carve((size_t)512 * 1024 * 2);
  h16* aW0p  = (h16*)carve((size_t)320 * 1024 * 2);
  h16* aW1p  = (h16*)carve((size_t)512 * 1024 * 2);
  h16* fWp   = (h16*)carve((size_t)768 * NOUTP * 2);

  auto pack = [&](const float* s, h16* d, int K, int N, int Kp, int Np) {
    int tot = Kp * Np;
    k_pack<<<(tot + 255) / 256, 256, 0, stream>>>(s, d, K, N, Kp, Np);
  };
  pack(emb_W, embWp, 350, 128, EKP, 128);
  pack(sW0, sW0p, 384, 1024, 384, 1024);
  pack(sW1, sW1p, 512, 1024, 512, 1024);
  pack(bW0, bW0p, 384, 1024, 384, 1024);
  pack(bW1, bW1p, 512, 1024, 512, 1024);
  pack(aW0, aW0p, 306, 1024, 320, 1024);
  pack(aW1, aW1p, 512, 1024, 512, 1024);
  pack(fW,  fWp,  768, NOUT, 768, NOUTP);

  // 106496 tokens / 16 per wave / 4 waves per block = 1664 blocks
  k_embed<<<1664, 128, 0, stream>>>(bw, bp, cw, cp, w_emb, p_emb, embWp, emb_b,
                                    buffEmb, compEmb);
  // 6144 sequences / 4 waves per block
  k_compose<<<1536, 128, 0, stream>>>(cw, ca, cal, compEmb, comp_a_emb,
                                      rec_W, rec_b, stackEmb);

  size_t smem384 = (size_t)16 * (128 + UU) * 2 + 16 * 512 * 2 + 16 * 1024 * 4;
  size_t smem320 = (size_t)16 * (64 + UU) * 2 + 16 * 512 * 2 + 16 * 1024 * 4;
  k_lstm<0, 128><<<8, 256, smem384, stream>>>(stackEmb, nullptr, nullptr, STK,
                                              sW0p, sb0, sW1p, sb1, slen, hS);
  k_lstm<1, 128><<<8, 256, smem384, stream>>>(buffEmb, nullptr, nullptr, BUFF,
                                              bW0p, bb0, bW1p, bb1, blen, hB);
  k_lstm<2, 64><<<8, 256, smem320, stream>>>(nullptr, ha, hist_a_emb, HIST,
                                             aW0p, ab0, aW1p, ab1, alen, hA);

  k_final<<<8, 32, 0, stream>>>(hS, hB, hA, fWp, fb, out);
}
